// HierarchicalModalityUntiedAttention_26525718020223
// MI455X (gfx1250) — matrix operand order repaired and checked by reading, not compiled, not run on hardware
//
#include <hip/hip_runtime.h>
#include <math.h>
#include <stdint.h>

typedef _Float16 v16h __attribute__((ext_vector_type(16)));
typedef _Float16 h8   __attribute__((ext_vector_type(8)));
typedef float    v8f  __attribute__((ext_vector_type(8)));

// Problem constants (from reference): B,S,D,HD,H = 4,2048,1024,64,16 ; E=H*HD=1024
#define Bb 4
#define Ss 2048
#define ShH 1024          // S/2
#define Dd 1024
#define Ee 1024
#define Hh 16
#define HDim 64
#define FFdim 4096
static constexpr float kInvSqrtHD = 0.125f;   // 1/sqrt(64)

// ---------------------------------------------------------------------------
// CDNA5 async memory->LDS DMA (ASYNCcnt-tracked), per CDNA5 ISA + 07:
//   global_load_async_to_lds_b128  vdst(=LDS byte offset), vaddr(64b), off
// Generic LDS pointers carry the wave-relative LDS offset in their low 32 bits.
// ---------------------------------------------------------------------------
__device__ __forceinline__ void async_copy_b128(const _Float16* g, _Float16* l) {
  const unsigned lds_off = (unsigned)(uintptr_t)l;
  asm volatile("global_load_async_to_lds_b128 %0, %1, off"
               :: "v"(lds_off), "v"(g)
               : "memory");
}
__device__ __forceinline__ void wait_async0() {
  asm volatile("s_wait_asynccnt 0x0" ::: "memory");
}
__device__ __forceinline__ void wait_async4() {
  asm volatile("s_wait_asynccnt 0x4" ::: "memory");
}

// ---------------------------------------------------------------------------
// WMMA fragment helpers (CDNA5 16x16x32 f16, wave32)
// A layout (16x32, 16-bit): lane L: M=L%16; half h -> K=h+16*(L/16)
// B layout (32x16, 16-bit): lane L: N=L%16; half h -> K=h+16*(L/16)
// C/D layout (16x16 f32):   lane L: N=L%16; vgpr r -> M=r+8*(L/16)
// ---------------------------------------------------------------------------
__device__ __forceinline__ v8f vzero8() {
  v8f z;
#pragma unroll
  for (int i = 0; i < 8; ++i) z[i] = 0.0f;
  return z;
}

__device__ __forceinline__ v16h load_fragA(const _Float16* __restrict__ base, int stride,
                                           int row0, int kbase, int lane) {
  const int r = row0 + (lane & 15);
  const int c = ((lane >> 4) << 4) + kbase;
  const _Float16* p = base + (long)r * stride + c;
  h8 lo = *(const h8*)p;          // K = c .. c+7
  h8 hi = *(const h8*)(p + 8);    // K = c+8 .. c+15
  v16h f;
#pragma unroll
  for (int i = 0; i < 8; ++i) { f[i] = lo[i]; f[i + 8] = hi[i]; }
  return f;
}

__device__ __forceinline__ v16h load_fragB(const _Float16* __restrict__ base, int stride,
                                           int row0, int kbase, int lane) {
  const int r = row0 + (lane & 15);            // N (output column) -> row of W / K-matrix
  const int c = ((lane >> 4) << 4) + kbase;    // 16 contiguous K values per lane
  const _Float16* p = base + (long)r * stride + c;
  h8 lo = *(const h8*)p;
  h8 hi = *(const h8*)(p + 8);
  v16h f;
#pragma unroll
  for (int i = 0; i < 8; ++i) { f[i] = lo[i]; f[i + 8] = hi[i]; }
  return f;
}

__device__ __forceinline__ v8f wmma32(v16h a, v16h b, v8f c) {
  return __builtin_amdgcn_wmma_f32_16x16x32_f16(false, a, false, b, (short)0, c, false, false);
}

__device__ __forceinline__ float rmax16(float v) {
  v = fmaxf(v, __shfl_xor(v, 1, 16));
  v = fmaxf(v, __shfl_xor(v, 2, 16));
  v = fmaxf(v, __shfl_xor(v, 4, 16));
  v = fmaxf(v, __shfl_xor(v, 8, 16));
  return v;
}
__device__ __forceinline__ float rsum16(float v) {
  v += __shfl_xor(v, 1, 16);
  v += __shfl_xor(v, 2, 16);
  v += __shfl_xor(v, 4, 16);
  v += __shfl_xor(v, 8, 16);
  return v;
}

// ---------------------------------------------------------------------------
// GEMM: C[M,N] = A[M,K] (f16) * W[N,K]^T (f16) + bias[N]; optional GELU.
// Block tile 128x128, 8 waves (4 along M x 2 along N), wave tile 32x64:
// 8 WMMA per 32-wide k-step against 6 fragment loads. LDS tiles are
// double-buffered and filled by global_load_async_to_lds_b128; the software
// pipeline issues tile k+1 then s_wait_asynccnt 4 to know tile k landed
// (per-wave in-order async completion).
// ---------------------------------------------------------------------------
template<bool BIAS, bool GELU, bool WF32, bool WF16>
__global__ void __launch_bounds__(256)
gemm_wmma_kernel(const _Float16* __restrict__ A, const _Float16* __restrict__ W,
                 const float* __restrict__ bias, float* __restrict__ Cf,
                 _Float16* __restrict__ Ch, int M, int N, int K) {
  __shared__ __align__(16) _Float16 As[2][128 * 40];
  __shared__ __align__(16) _Float16 Bs[2][128 * 40];
  const int tid  = threadIdx.x;
  const int lane = tid & 31;
  const int wid  = tid >> 5;
  const int wm   = wid & 3;   // 0..3 along M (32 rows each)
  const int wn   = wid >> 2;  // 0..1 along N (64 cols each)
  const long bm  = (long)blockIdx.x * 128;
  const long bn  = (long)blockIdx.y * 128;

  const int lr = tid >> 2;        // 0..63
  const int lc = (tid & 3) * 8;   // 0,8,16,24

  const _Float16* gA0 = A + (bm + lr)      * (long)K + lc;
  const _Float16* gA1 = A + (bm + lr + 64) * (long)K + lc;
  const _Float16* gW0 = W + (bn + lr)      * (long)K + lc;
  const _Float16* gW1 = W + (bn + lr + 64) * (long)K + lc;

  auto issue_tile = [&](int k0, int buf) {
    async_copy_b128(gA0 + k0, &As[buf][lr * 40 + lc]);
    async_copy_b128(gA1 + k0, &As[buf][(lr + 64) * 40 + lc]);
    async_copy_b128(gW0 + k0, &Bs[buf][lr * 40 + lc]);
    async_copy_b128(gW1 + k0, &Bs[buf][(lr + 64) * 40 + lc]);
  };

  v8f acc[2][4];
#pragma unroll
  for (int i = 0; i < 2; ++i)
#pragma unroll
    for (int j = 0; j < 4; ++j) acc[i][j] = vzero8();

  const int nsteps = K >> 5;
  issue_tile(0, 0);
  for (int s = 0; s < nsteps; ++s) {
    const int buf = s & 1;
    if (s + 1 < nsteps) {
      issue_tile((s + 1) << 5, buf ^ 1);
      wait_async4();              // tile s complete, tile s+1 in flight
    } else {
      wait_async0();
    }
    __syncthreads();

    v16h a0 = load_fragA(As[buf], 40, wm * 32,      0, lane);
    v16h a1 = load_fragA(As[buf], 40, wm * 32 + 16, 0, lane);
    v16h b0 = load_fragB(Bs[buf], 40, wn * 64,      0, lane);
    v16h b1 = load_fragB(Bs[buf], 40, wn * 64 + 16, 0, lane);
    v16h b2 = load_fragB(Bs[buf], 40, wn * 64 + 32, 0, lane);
    v16h b3 = load_fragB(Bs[buf], 40, wn * 64 + 48, 0, lane);
    acc[0][0] = wmma32(a0, b0, acc[0][0]);
    acc[0][1] = wmma32(a0, b1, acc[0][1]);
    acc[0][2] = wmma32(a0, b2, acc[0][2]);
    acc[0][3] = wmma32(a0, b3, acc[0][3]);
    acc[1][0] = wmma32(a1, b0, acc[1][0]);
    acc[1][1] = wmma32(a1, b1, acc[1][1]);
    acc[1][2] = wmma32(a1, b2, acc[1][2]);
    acc[1][3] = wmma32(a1, b3, acc[1][3]);
    __syncthreads();
  }

#pragma unroll
  for (int i = 0; i < 2; ++i)
#pragma unroll
    for (int j = 0; j < 4; ++j) {
      const int nloc = wn * 64 + j * 16 + (lane & 15);
      const long n   = bn + nloc;
      const float bv = BIAS ? bias[n] : 0.0f;
#pragma unroll
      for (int r = 0; r < 8; ++r) {
        const long m = bm + wm * 32 + i * 16 + r + 8 * (lane >> 4);
        float v = acc[i][j][r] + bv;
        if (GELU) v = 0.5f * v * (1.0f + erff(v * 0.70710678118654752f));
        const long off = m * (long)N + n;
        if (WF32) Cf[off] = v;
        if (WF16) Ch[off] = (_Float16)v;
      }
    }
}

// ---------------------------------------------------------------------------
// Flash attention: one wave per (b, h, 16-row q tile). Online softmax over
// 32-key chunks. QK^T and PV both via v_wmma_f32_16x16x32_f16. The V chunk is
// DMA'd to LDS with async-to-LDS ops that overlap the QK^T WMMAs; the P tile
// round-trips through LDS to convert C/D layout -> A-fragment layout.
// qp/kp/vp layout: [B, L, H, HD] f16 (row stride E).
// ---------------------------------------------------------------------------
__global__ void __launch_bounds__(32)
flash_attn_kernel(const _Float16* __restrict__ Q, const _Float16* __restrict__ Kp,
                  const _Float16* __restrict__ Vp, _Float16* __restrict__ O,
                  int L, float scale) {
  __shared__ __align__(16) _Float16 Vs[32 * 72];  // V chunk, row-major, padded
  __shared__ __align__(16) _Float16 Ps[16 * 40];  // P tile re-layout buffer
  const int lane = threadIdx.x;
  const int q0   = blockIdx.x * 16;
  const int h    = blockIdx.y;
  const int b    = blockIdx.z;
  const long base_bh = ((long)b * L) * Ee + (long)h * HDim;
  const _Float16* qrow = Q + base_bh + (long)q0 * Ee;

  // Q fragments (rows q0..q0+15, HD split 0..31 / 32..63), persistent
  v16h qa0 = load_fragA(qrow, Ee, 0, 0,  lane);
  v16h qa1 = load_fragA(qrow, Ee, 0, 32, lane);

  float mrow[8], lrow[8];
  v8f acc[4];
#pragma unroll
  for (int r = 0; r < 8; ++r) { mrow[r] = -1e30f; lrow[r] = 0.0f; }
#pragma unroll
  for (int d = 0; d < 4; ++d) acc[d] = vzero8();

  for (int kb = 0; kb < L; kb += 32) {
    // async-DMA 32 V rows (64 halves each) into LDS; overlaps QK^T below
    const _Float16* vrow = Vp + base_bh + (long)(kb + lane) * Ee;
#pragma unroll
    for (int i = 0; i < 8; ++i)
      async_copy_b128(vrow + i * 8, &Vs[lane * 72 + i * 8]);

    // S = Q * K^T for keys kb..kb+31 (B operand = rows of K, contiguous)
    const _Float16* krow = Kp + base_bh + (long)kb * Ee;
    v16h k00 = load_fragB(krow, Ee, 0,  0,  lane);
    v16h k01 = load_fragB(krow, Ee, 0,  32, lane);
    v16h k10 = load_fragB(krow, Ee, 16, 0,  lane);
    v16h k11 = load_fragB(krow, Ee, 16, 32, lane);
    v8f s0 = vzero8(), s1 = vzero8();
    s0 = wmma32(qa0, k00, s0); s0 = wmma32(qa1, k01, s0);
    s1 = wmma32(qa0, k10, s1); s1 = wmma32(qa1, k11, s1);

    // online softmax; lane holds col (lane&15), rows r+8*(lane>>4)
#pragma unroll
    for (int r = 0; r < 8; ++r) {
      const float v0 = s0[r] * scale;
      const float v1 = s1[r] * scale;
      const float vmax = rmax16(fmaxf(v0, v1));
      const float mn = fmaxf(mrow[r], vmax);
      const float alpha = __expf(mrow[r] - mn);
      const float p0 = __expf(v0 - mn);
      const float p1 = __expf(v1 - mn);
      lrow[r] = lrow[r] * alpha + rsum16(p0 + p1);
      mrow[r] = mn;
#pragma unroll
      for (int d = 0; d < 4; ++d) acc[d][r] *= alpha;
      const int prow = r + 8 * (lane >> 4);
      Ps[prow * 40 + (lane & 15)]      = (_Float16)p0;
      Ps[prow * 40 + (lane & 15) + 16] = (_Float16)p1;
    }
    wait_async0();      // V chunk resident in LDS
    __syncthreads();    // P tile visible (DScnt drained)

    // O += P(16x32) * V(32x64): A-frag from Ps, B-frags gathered from Vs
    v16h pa = load_fragA(Ps, 40, 0, 0, lane);
#pragma unroll
    for (int d = 0; d < 4; ++d) {
      v16h vb;
      const int n = d * 16 + (lane & 15);
      const int c = (lane >> 4) * 16;
#pragma unroll
      for (int i = 0; i < 16; ++i) vb[i] = Vs[(c + i) * 72 + n];
      acc[d] = wmma32(pa, vb, acc[d]);
    }
    __syncthreads();
  }

#pragma unroll
  for (int r = 0; r < 8; ++r) {
    const float invl = 1.0f / lrow[r];
    const long row = q0 + r + 8 * (lane >> 4);
    const long obase = base_bh + row * (long)Ee;
#pragma unroll
    for (int d = 0; d < 4; ++d)
      O[obase + d * 16 + (lane & 15)] = (_Float16)(acc[d][r] * invl);
  }
}

// ---------------------------------------------------------------------------
// Elementwise / layout kernels
// ---------------------------------------------------------------------------
__global__ void cvt_f32_f16_kernel(_Float16* __restrict__ dst,
                                   const float* __restrict__ src, long n) {
  const long i = (long)blockIdx.x * blockDim.x + threadIdx.x;
  if (i < n) dst[i] = (_Float16)src[i];
}

__global__ void __launch_bounds__(256)
gather_cvt_kernel(_Float16* __restrict__ dst, const float* __restrict__ src,
                  const int* __restrict__ idx, int rowsPerB, int S) {
  const int row = blockIdx.x;
  const int b = row / rowsPerB, s = row % rowsPerB;
  const int tok = idx[s];
  const long sbase = ((long)b * S + tok) * Dd;
  const long dbase = (long)row * Dd;
#pragma unroll
  for (int k = 0; k < 4; ++k) {
    const int c = threadIdx.x + k * 256;
    dst[dbase + c] = (_Float16)src[sbase + c];
  }
}

__global__ void __launch_bounds__(256)
gather_f16_kernel(_Float16* __restrict__ dst, const _Float16* __restrict__ src,
                  const int* __restrict__ idx, int rowsPerB, int S) {
  const int row = blockIdx.x;
  const int b = row / rowsPerB, s = row % rowsPerB;
  const int tok = idx[s];
  const long sbase = ((long)b * S + tok) * Ee;
  const long dbase = (long)row * Ee;
#pragma unroll
  for (int k = 0; k < 4; ++k) {
    const int c = threadIdx.x + k * 256;
    dst[dbase + c] = src[sbase + c];
  }
}

// out[b, idx[s], :] = resid[b, idx[s], :] + LN(src[b*rowsPerB+s, :]) * g + be
__global__ void __launch_bounds__(256)
ln_residual_kernel(const float* __restrict__ src, const float* __restrict__ residBase,
                   const int* __restrict__ idx, const float* __restrict__ g,
                   const float* __restrict__ be, float* __restrict__ outF,
                   _Float16* __restrict__ outH, int rowsPerB, int S) {
  __shared__ float red[256];
  const int tid = threadIdx.x;
  const int row = blockIdx.x;
  const int b = row / rowsPerB, s = row % rowsPerB;
  const int tok = idx[s];
  const float* xr = src + (long)row * Dd;
  float xv[4];
  float loc = 0.0f;
#pragma unroll
  for (int k = 0; k < 4; ++k) { xv[k] = xr[tid + k * 256]; loc += xv[k]; }
  red[tid] = loc; __syncthreads();
  for (int st = 128; st > 0; st >>= 1) {
    if (tid < st) red[tid] += red[tid + st];
    __syncthreads();
  }
  const float mean = red[0] * (1.0f / Dd);
  __syncthreads();
  loc = 0.0f;
#pragma unroll
  for (int k = 0; k < 4; ++k) { const float d = xv[k] - mean; loc += d * d; }
  red[tid] = loc; __syncthreads();
  for (int st = 128; st > 0; st >>= 1) {
    if (tid < st) red[tid] += red[tid + st];
    __syncthreads();
  }
  const float rstd = rsqrtf(red[0] * (1.0f / Dd) + 1e-5f);
  const long obase = ((long)b * S + tok) * Dd;
#pragma unroll
  for (int k = 0; k < 4; ++k) {
    const int c = tid + k * 256;
    const float v = (xv[k] - mean) * rstd * g[c] + be[c] + residBase[obase + c];
    if (outF) outF[obase + c] = v;
    if (outH) outH[obase + c] = (_Float16)v;
  }
}

// ---------------------------------------------------------------------------
// Host orchestration
// ---------------------------------------------------------------------------
extern "C" void kernel_launch(void* const* d_in, const int* in_sizes, int n_in,
                              void* d_out, int out_size, void* d_ws, size_t ws_size,
                              hipStream_t stream) {
  (void)in_sizes; (void)n_in; (void)out_size; (void)ws_size;
  // setup_inputs order: x, params{cwq0,cwq1,cwk0,cwk1,cwv0,cwv1, cross_in_w,
  // cross_in_b, cross_out_w, cross_out_b, cn0_g, cn0_b, cn1_g, cn1_b,
  // uni_in_w, uni_in_b, uni_out_w, uni_out_b, ffn_w1, ffn_b1, ffn_w2, ffn_b2,
  // ffn_ln_g, ffn_ln_b}, attn_mask (unused), mask0_idx, mask1_idx
  const float* x    = (const float*)d_in[0];
  const float* cw[6];
  for (int i = 0; i < 6; ++i) cw[i] = (const float*)d_in[1 + i]; // q0,q1,k0,k1,v0,v1
  const float* ciw  = (const float*)d_in[7];
  const float* cib  = (const float*)d_in[8];
  const float* cow  = (const float*)d_in[9];
  const float* cob  = (const float*)d_in[10];
  const float* cng[2] = {(const float*)d_in[11], (const float*)d_in[13]};
  const float* cnb[2] = {(const float*)d_in[12], (const float*)d_in[14]};
  const float* uiw  = (const float*)d_in[15];
  const float* uib  = (const float*)d_in[16];
  const float* uow  = (const float*)d_in[17];
  const float* uob  = (const float*)d_in[18];
  const float* fw1  = (const float*)d_in[19];
  const float* fb1  = (const float*)d_in[20];
  const float* fw2  = (const float*)d_in[21];
  const float* fb2  = (const float*)d_in[22];
  const float* flg  = (const float*)d_in[23];
  const float* flb  = (const float*)d_in[24];
  const int* idxs[2] = {(const int*)d_in[26], (const int*)d_in[27]};
  float* out = (float*)d_out;

  const size_t ED = (size_t)Ee * Dd;          // 1M
  const int Mc = Bb * ShH;                    // 4096 rows (half)
  const int Mu = Bb * Ss;                     // 8192 rows (full)

  char* ws = (char*)d_ws;
  size_t off = 0;
  auto alloc = [&](size_t bytes) -> void* {
    off = (off + 255) & ~(size_t)255;
    void* p = ws + off;
    off += bytes;
    return p;
  };

  // f16 weight copies
  _Float16* cw16[6];
  for (int i = 0; i < 6; ++i) cw16[i] = (_Float16*)alloc(ED * 2);
  _Float16* ciw16 = (_Float16*)alloc(3 * ED * 2);
  _Float16* cow16 = (_Float16*)alloc(ED * 2);
  _Float16* uiw16 = (_Float16*)alloc(3 * ED * 2);
  _Float16* uow16 = (_Float16*)alloc(ED * 2);
  _Float16* fw116 = (_Float16*)alloc(4 * ED * 2);
  _Float16* fw216 = (_Float16*)alloc(4 * ED * 2);
  // activations
  _Float16* toks16[2] = {(_Float16*)alloc((size_t)Mc * Dd * 2),
                         (_Float16*)alloc((size_t)Mc * Dd * 2)};
  _Float16* qlin = (_Float16*)alloc((size_t)Mc * Ee * 2);
  _Float16* klin = (_Float16*)alloc((size_t)Mc * Ee * 2);
  _Float16* vlin = (_Float16*)alloc((size_t)Mc * Ee * 2);
  _Float16* qp   = (_Float16*)alloc((size_t)Mc * Ee * 2);
  _Float16* kp   = (_Float16*)alloc((size_t)Mc * Ee * 2);
  _Float16* vp   = (_Float16*)alloc((size_t)Mc * Ee * 2);
  _Float16* oc   = (_Float16*)alloc((size_t)Mc * Ee * 2);
  float*    cm   = (float*)   alloc((size_t)Mc * Ee * 4);
  _Float16* enh16= (_Float16*)alloc((size_t)Mu * Ee * 2);
  _Float16* qpu  = (_Float16*)alloc((size_t)Mu * Ee * 2);
  _Float16* kpu  = (_Float16*)alloc((size_t)Mu * Ee * 2);
  _Float16* vpu  = (_Float16*)alloc((size_t)Mu * Ee * 2);
  _Float16* ou   = (_Float16*)alloc((size_t)Mu * Ee * 2);
  _Float16* out16= (_Float16*)alloc((size_t)Mu * Ee * 2);
  _Float16* mt16 = (_Float16*)alloc((size_t)Mc * Ee * 2);
  _Float16* h16  = (_Float16*)alloc((size_t)Mc * FFdim * 2);
  float*    ffn  = (float*)   alloc((size_t)Mc * Dd * 4);

  auto cvt = [&](_Float16* dst, const float* src, long n) {
    cvt_f32_f16_kernel<<<(unsigned)((n + 255) / 256), 256, 0, stream>>>(dst, src, n);
  };
  for (int i = 0; i < 6; ++i) cvt(cw16[i], cw[i], (long)ED);
  cvt(ciw16, ciw, 3 * (long)ED);
  cvt(cow16, cow, (long)ED);
  cvt(uiw16, uiw, 3 * (long)ED);
  cvt(uow16, uow, (long)ED);
  cvt(fw116, fw1, 4 * (long)ED);
  cvt(fw216, fw2, 4 * (long)ED);
  cvt(enh16, x, (long)Mu * Dd);   // enhanced initialized to x (f16 for GEMMs)

  gather_cvt_kernel<<<Mc, 256, 0, stream>>>(toks16[0], x, idxs[0], ShH, Ss);
  gather_cvt_kernel<<<Mc, 256, 0, stream>>>(toks16[1], x, idxs[1], ShH, Ss);

  const dim3 gC(Mc / 128, Ee / 128);
  // ---- cross-attention, both directions ----
  for (int i = 0; i < 2; ++i) {
    const int j = 1 - i;
    gemm_wmma_kernel<false, false, false, true><<<gC, 256, 0, stream>>>(
        toks16[i], cw16[0 + i], nullptr, nullptr, qlin, Mc, Ee, Dd);       // cwq_i
    gemm_wmma_kernel<false, false, false, true><<<gC, 256, 0, stream>>>(
        toks16[j], cw16[2 + j], nullptr, nullptr, klin, Mc, Ee, Dd);       // cwk_j
    gemm_wmma_kernel<false, false, false, true><<<gC, 256, 0, stream>>>(
        toks16[j], cw16[4 + j], nullptr, nullptr, vlin, Mc, Ee, Dd);       // cwv_j
    gemm_wmma_kernel<true, false, false, true><<<gC, 256, 0, stream>>>(
        qlin, ciw16,             cib,          nullptr, qp, Mc, Ee, Ee);
    gemm_wmma_kernel<true, false, false, true><<<gC, 256, 0, stream>>>(
        klin, ciw16 + ED,        cib + Ee,     nullptr, kp, Mc, Ee, Ee);
    gemm_wmma_kernel<true, false, false, true><<<gC, 256, 0, stream>>>(
        vlin, ciw16 + 2 * ED,    cib + 2 * Ee, nullptr, vp, Mc, Ee, Ee);
    flash_attn_kernel<<<dim3(ShH / 16, Hh, Bb), 32, 0, stream>>>(
        qp, kp, vp, oc, ShH, kInvSqrtHD);
    gemm_wmma_kernel<true, false, true, false><<<gC, 256, 0, stream>>>(
        oc, cow16, cob, cm, nullptr, Mc, Ee, Ee);
    ln_residual_kernel<<<Mc, 256, 0, stream>>>(
        cm, x, idxs[i], cng[i], cnb[i], nullptr, enh16, ShH, Ss);
  }

  // ---- full self-attention on enhanced ----
  const dim3 gU(Mu / 128, Ee / 128);
  gemm_wmma_kernel<true, false, false, true><<<gU, 256, 0, stream>>>(
      enh16, uiw16,          uib,          nullptr, qpu, Mu, Ee, Ee);
  gemm_wmma_kernel<true, false, false, true><<<gU, 256, 0, stream>>>(
      enh16, uiw16 + ED,     uib + Ee,     nullptr, kpu, Mu, Ee, Ee);
  gemm_wmma_kernel<true, false, false, true><<<gU, 256, 0, stream>>>(
      enh16, uiw16 + 2 * ED, uib + 2 * Ee, nullptr, vpu, Mu, Ee, Ee);
  flash_attn_kernel<<<dim3(Ss / 16, Hh, Bb), 32, 0, stream>>>(
      qpu, kpu, vpu, ou, Ss, kInvSqrtHD);
  gemm_wmma_kernel<true, false, true, true><<<gU, 256, 0, stream>>>(
      ou, uow16, uob, out, out16, Mu, Ee, Ee);

  // ---- FFN on mask1 half, residual LN back into d_out ----
  gather_f16_kernel<<<Mc, 256, 0, stream>>>(mt16, out16, idxs[1], ShH, Ss);
  gemm_wmma_kernel<true, true, false, true><<<dim3(Mc / 128, FFdim / 128), 256, 0, stream>>>(
      mt16, fw116, fb1, nullptr, h16, Mc, FFdim, Dd);
  gemm_wmma_kernel<true, false, true, false><<<dim3(Mc / 128, Dd / 128), 256, 0, stream>>>(
      h16, fw216, fb2, ffn, nullptr, Mc, Dd, FFdim);
  ln_residual_kernel<<<Mc, 256, 0, stream>>>(
      ffn, out, idxs[1], flg, flb, out, nullptr, ShH, Ss);
}
